// RNN_25589415149939
// MI455X (gfx1250) — matrix operand order repaired and checked by reading, not compiled, not run on hardware
//
#include <hip/hip_runtime.h>

// ---------------------------------------------------------------------------
// CDNA5 (gfx1250) implementation of the DyGCRN video-prediction reference.
// All heavy math on v_wmma_f32_16x16x32_f16 (wave32, one wave per 16x16 tile).
// A-fragments: two contiguous b128 loads per lane.  B-fragments: weights are
// pre-packed into fragment-major layout -> one contiguous 32B load per lane.
// ---------------------------------------------------------------------------

typedef _Float16 half_t;
typedef __attribute__((ext_vector_type(16))) _Float16 v16h;
typedef __attribute__((ext_vector_type(8)))  _Float16 v8h;
typedef __attribute__((ext_vector_type(8)))  float    v8f;

#define DEVINL __device__ __forceinline__

DEVINL v8f wmma16(v16h a, v16h b, v8f c) {
  return __builtin_amdgcn_wmma_f32_16x16x32_f16(false, a, false, b, (short)0, c,
                                                false, false);
}

// A 16x32 f16 fragment (contiguous k, matches packed B): lane's 16 halfs are
// row[hh*16..+15]  -> two unconditional 16B loads.
DEVINL v16h load_a_full(const half_t* row, int hh) {
  v8h lo = *(const v8h*)(row + (hh << 4));
  v8h hi = *(const v8h*)(row + (hh << 4) + 8);
  v16h a;
#pragma unroll
  for (int e = 0; e < 8; ++e) { a[e] = lo[e]; a[8 + e] = hi[e]; }
  return a;
}

// branch-free zeroing (lowers to v_cndmask, no exec manipulation)
DEVINL v16h zero_unless(v16h a, bool keep) {
#pragma unroll
  for (int e = 0; e < 16; ++e) a[e] = keep ? a[e] : (half_t)0.f;
  return a;
}

// B 32x16 f16 fragment from fragment-major packed storage:
// packed[tile][lane][e] = B[hh*16+e][nl]  -> one contiguous 32B load per lane.
DEVINL v16h load_b_packed(const half_t* tileBase, int lane) {
  return *(const v16h*)(tileBase + (lane << 4));
}

DEVINL float actf(float x, int act) {
  if (act == 1) return fmaxf(x, 0.f);
  if (act == 2) return 1.f / (1.f + __expf(-x));
  if (act == 3) return tanhf(x);
  return x;
}
DEVINL int imin(int a, int b) { return a < b ? a : b; }
DEVINL int imax(int a, int b) { return a > b ? a : b; }

// --------------------------- generic WMMA GEMM ------------------------------
// C[m,n] = act(sum_k A[m,k]*Bp[k,n] + bias[n]); A row-major, B fragment-packed
// with tile order [kt][nt].  One wave per 16x16 output tile.
__global__ void k_gemm(const half_t* __restrict__ A, const half_t* __restrict__ Bp,
                       const float* __restrict__ bias, void* __restrict__ C,
                       int N, int K, int lda, int ldc,
                       long sA, long sB, long sC, int act, int outHalf) {
  int lane = threadIdx.x & 31;
  int hh = lane >> 4, nl = lane & 15;
  int m0 = blockIdx.x << 4, nt = blockIdx.y;
  int NT = N >> 4;
  const half_t* arow = A + sA * blockIdx.z + (long)(m0 + nl) * lda;
  const half_t* bp = Bp + sB * blockIdx.z + (long)nt * 512 + (lane << 4);
  long bstep = (long)NT * 512;
  v8f acc = {};
  int KT = K >> 5;
  for (int kt = 0; kt < KT; ++kt) {
    v16h a = load_a_full(arow + (kt << 5), hh);
    v16h b = *(const v16h*)(bp + kt * bstep);
    acc = wmma16(a, b, acc);
  }
  int n = (nt << 4) + nl;
  float bv = bias ? bias[n] : 0.f;
#pragma unroll
  for (int v = 0; v < 8; ++v) {
    int m = m0 + v + (hh << 3);
    float x = actf(acc[v] + bv, act);
    long idx = sC * blockIdx.z + (long)m * ldc + n;
    if (outHalf) ((half_t*)C)[idx] = (half_t)x;
    else         ((float*)C)[idx] = x;
  }
}

// ------------- per-node graph conv GEMM (AGCRN weight pool) -----------------
// out[b,node,o] = act( sum_{cheb,k} xg[cheb][b,node,k]*Wn[node,cheb,k,o] + Eb )
// M = batch = 16 (exactly one WMMA tile), K = Ci, per-node packed B.
__global__ void k_gconv(const half_t* __restrict__ xg0, const half_t* __restrict__ xg1,
                        const half_t* __restrict__ Wn, const float* __restrict__ Eb,
                        half_t* __restrict__ out, int Ci, int Co, int act) {
  int lane = threadIdx.x & 31;
  int hh = lane >> 4, nl = lane & 15;
  int node = blockIdx.x;
  int nt = blockIdx.y;
  int KT = Ci >> 5, NT = Co >> 4;
  v8f acc = {};
#pragma unroll
  for (int cheb = 0; cheb < 2; ++cheb) {
    const half_t* xg = cheb ? xg1 : xg0;
    const half_t* arow = xg + ((long)nl * 256 + node) * Ci;
    const half_t* bp = Wn + ((((long)node * 2 + cheb) * KT) * NT + nt) * 512 + (lane << 4);
    for (int kt = 0; kt < KT; ++kt) {
      v16h a = load_a_full(arow + (kt << 5), hh);
      v16h b = *(const v16h*)(bp + (long)kt * NT * 512);
      acc = wmma16(a, b, acc);
    }
  }
  int o = (nt << 4) + nl;
  float bv = Eb[node * Co + o];
#pragma unroll
  for (int v = 0; v < 8; ++v) {
    int bb = v + (hh << 3);  // batch row
    float x = actf(acc[v] + bv, act);
    out[((long)bb * 256 + node) * Co + o] = (half_t)x;
  }
}

// ----------------- 3x3 SAME conv as implicit WMMA GEMM ----------------------
// in: NHWC f16 [Nimg,H,W,Ci]; wt packed per (tap,kt,nt) tile (zero-padded K);
// out raw f32 NHWC.  Border pixels: clamped address + cndmask zeroing.
__global__ void k_conv3(const half_t* __restrict__ in, const half_t* __restrict__ wt,
                        float* __restrict__ out, int Nimg, int H, int W, int Ci, int Co) {
  int lane = threadIdx.x & 31;
  int hh = lane >> 4, nl = lane & 15;
  int wtiles = W >> 4;
  int bx = blockIdx.x;
  int w0 = (bx % wtiles) << 4;
  int rest = bx / wtiles;
  int h = rest % H;
  int img = rest / H;
  int nt = blockIdx.y;
  int KT = (Ci + 31) >> 5, NT = Co >> 4;
  int wp = w0 + nl;  // pixel column owned by this lane's A row
  v8f acc = {};
  for (int dy = -1; dy <= 1; ++dy) {
    int hs = h + dy;
    if (hs < 0 || hs >= H) continue;  // wave-uniform
    for (int dx = -1; dx <= 1; ++dx) {
      int ws = wp + dx;
      bool pv = (ws >= 0) && (ws < W);
      int wsc = imin(imax(ws, 0), W - 1);  // clamped, always-valid address
      const half_t* arow = in + (((long)img * H + hs) * W + wsc) * Ci;
      int tap = (dy + 1) * 3 + (dx + 1);
      const half_t* bp = wt + (((long)tap * KT) * NT + nt) * 512 + (lane << 4);
      for (int kt = 0; kt < KT; ++kt) {
        v16h a = zero_unless(load_a_full(arow + (kt << 5), hh), pv);
        v16h b = *(const v16h*)(bp + (long)kt * NT * 512);
        acc = wmma16(a, b, acc);
      }
    }
  }
#pragma unroll
  for (int v = 0; v < 8; ++v) {
    int m = v + (hh << 3);
    out[(((long)img * H + h) * W + w0 + m) * Co + (nt << 4) + nl] = acc[v];
  }
}

// ------------- ODE second GEMM with fused Euler update ----------------------
// s += dt*(hidden@w2 + b2); emits f32 state, f16 state copy, f16 trajectory.
__global__ void k_ode2(const half_t* __restrict__ hid, const half_t* __restrict__ w2p,
                       const float* __restrict__ b2, float* __restrict__ s,
                       half_t* __restrict__ s16, half_t* __restrict__ trajt, float dt) {
  int lane = threadIdx.x & 31;
  int hh = lane >> 4, nl = lane & 15;
  int m0 = blockIdx.x << 4, nt = blockIdx.y;  // K=128 (KT=4), N=64 (NT=4)
  const half_t* arow = hid + (long)(m0 + nl) * 128;
  const half_t* bp = w2p + (long)nt * 512 + (lane << 4);
  v8f acc = {};
#pragma unroll
  for (int kt = 0; kt < 4; ++kt) {
    v16h a = load_a_full(arow + (kt << 5), hh);
    v16h b = *(const v16h*)(bp + (long)kt * 4 * 512);
    acc = wmma16(a, b, acc);
  }
  int n = (nt << 4) + nl;
  float bv = b2[n];
#pragma unroll
  for (int v = 0; v < 8; ++v) {
    int m = m0 + v + (hh << 3);
    long idx = (long)m * 64 + n;
    float sv = s[idx] + dt * (acc[v] + bv);
    s[idx] = sv;
    half_t hv = (half_t)sv;
    s16[idx] = hv;
    trajt[idx] = hv;
  }
}

// --------------------------- small helper kernels ---------------------------
#define GRID_STRIDE(i, total)                                                   \
  for (long i = blockIdx.x * (long)blockDim.x + threadIdx.x; i < (total);       \
       i += (long)gridDim.x * blockDim.x)

__global__ void k_f2h(const float* x, half_t* y, long n) {
  GRID_STRIDE(i, n) y[i] = (half_t)x[i];
}

// OIHW f32 3x3 conv weight -> fragment-packed [tap][kt][nt][lane][16] f16,
// zero-padded in K so conv A loads need no channel guard.
__global__ void k_pack3x3(const float* __restrict__ w, half_t* __restrict__ o,
                          int Co, int Ci) {
  int KT = (Ci + 31) >> 5, NT = Co >> 4;
  long total = 9L * KT * NT * 512;
  GRID_STRIDE(i, total) {
    int e = (int)(i & 15);
    int lane = (int)((i >> 4) & 31);
    long tile = i >> 9;
    int nt = (int)(tile % NT); tile /= NT;
    int kt = (int)(tile % KT);
    int tap = (int)(tile / KT);
    int k = (kt << 5) + ((lane >> 4) << 4) + e;
    int co = (nt << 4) + (lane & 15);
    int ky = tap / 3, kx = tap % 3;
    float v = (k < Ci) ? w[(((long)co * Ci + k) * 3 + ky) * 3 + kx] : 0.f;
    o[i] = (half_t)v;
  }
}

// [Co,Ci,1,1] f32 -> fragment-packed [kt][nt][lane][16] f16  (K=Ci, N=Co)
__global__ void k_pack1x1(const float* __restrict__ w, half_t* __restrict__ o,
                          int Co, int Ci) {
  int KT = Ci >> 5, NT = Co >> 4;
  long total = (long)KT * NT * 512;
  GRID_STRIDE(i, total) {
    int e = (int)(i & 15);
    int lane = (int)((i >> 4) & 31);
    long tile = i >> 9;
    int nt = (int)(tile % NT);
    int kt = (int)(tile / NT);
    int k = (kt << 5) + ((lane >> 4) << 4) + e;
    int co = (nt << 4) + (lane & 15);
    o[i] = (half_t)w[(long)co * Ci + k];
  }
}

// dense f32 [K,N] -> fragment-packed f16
__global__ void k_packB(const float* __restrict__ src, half_t* __restrict__ o,
                        int K, int N) {
  int KT = K >> 5, NT = N >> 4;
  long total = (long)KT * NT * 512;
  GRID_STRIDE(i, total) {
    int e = (int)(i & 15);
    int lane = (int)((i >> 4) & 31);
    long tile = i >> 9;
    int nt = (int)(tile % NT);
    int kt = (int)(tile / NT);
    int k = (kt << 5) + ((lane >> 4) << 4) + e;
    int n = (nt << 4) + (lane & 15);
    o[i] = (half_t)src[(long)k * N + n];
  }
}

// training-mode BN statistics over NHWC f32 raw conv output (one block/channel)
__global__ void k_bnstats(const float* __restrict__ x, const float* __restrict__ g,
                          const float* __restrict__ b, float* scale, float* shift,
                          long NP, int C) {
  int c = blockIdx.x;
  float s = 0.f, s2 = 0.f;
  for (long i = threadIdx.x; i < NP; i += blockDim.x) {
    float v = x[i * C + c];
    s += v; s2 += v * v;
  }
  __shared__ float r1[256], r2[256];
  int tid = threadIdx.x;
  r1[tid] = s; r2[tid] = s2;
  __syncthreads();
  for (int st = 128; st > 0; st >>= 1) {
    if (tid < st) { r1[tid] += r1[tid + st]; r2[tid] += r2[tid + st]; }
    __syncthreads();
  }
  if (tid == 0) {
    float m = r1[0] / (float)NP;
    float var = r2[0] / (float)NP - m * m;
    float sc = g[c] * rsqrtf(var + 1e-5f);
    scale[c] = sc;
    shift[c] = b[c] - m * sc;
  }
}

__global__ void k_bnact(const float* x, const float* scale, const float* shift,
                        half_t* y, long total, int C) {
  GRID_STRIDE(i, total) {
    int c = (int)(i % C);
    float v = x[i] * scale[c] + shift[c];
    y[i] = (half_t)fmaxf(v, 0.f);
  }
}

__global__ void k_maxpool(const half_t* in, half_t* out, int Nimg, int H, int W, int C) {
  int Ho = H >> 1, Wo = W >> 1;
  long total = (long)Nimg * Ho * Wo * C;
  GRID_STRIDE(i, total) {
    int c = (int)(i % C);
    long r = i / C;
    int wo = (int)(r % Wo); r /= Wo;
    int ho = (int)(r % Ho);
    long img = r / Ho;
    const half_t* p = in + (((img * H + ho * 2) * W) + wo * 2) * C + c;
    float a0 = (float)p[0], a1 = (float)p[C];
    float a2 = (float)p[(long)W * C], a3 = (float)p[(long)W * C + C];
    out[i] = (half_t)fmaxf(fmaxf(a0, a1), fmaxf(a2, a3));
  }
}

// E_dyn[n,:] = softmax_m(E[n,:]·M[m,:]) @ M   (one wave32 per node, MEM==32)
__global__ void k_edyn(const float* __restrict__ E, const float* __restrict__ Mm,
                       float* __restrict__ Edyn) {
  int n = blockIdx.x;
  int lane = threadIdx.x;
  float d = 0.f;
  for (int e = 0; e < 10; ++e) d += E[n * 10 + e] * Mm[lane * 10 + e];
  float mx = d;
  for (int o = 16; o > 0; o >>= 1) mx = fmaxf(mx, __shfl_xor(mx, o, 32));
  float ex = __expf(d - mx);
  float sm = ex;
  for (int o = 16; o > 0; o >>= 1) sm += __shfl_xor(sm, o, 32);
  float pm = ex / sm;
  for (int e = 0; e < 10; ++e) {
    float v = pm * Mm[lane * 10 + e];
    for (int o = 16; o > 0; o >>= 1) v += __shfl_xor(v, o, 32);
    if (lane == 0) Edyn[n * 10 + e] = v;
  }
}

// sup[n,:] = softmax(relu(Edyn[n,:]·Edyn[m,:]))   (one block of 256 per row)
__global__ void k_sup(const float* __restrict__ Edyn, half_t* __restrict__ sup) {
  int n = blockIdx.x, m = threadIdx.x;
  float d = 0.f;
  for (int e = 0; e < 10; ++e) d += Edyn[n * 10 + e] * Edyn[m * 10 + e];
  d = fmaxf(d, 0.f);
  __shared__ float red[256];
  red[m] = d; __syncthreads();
  for (int st = 128; st > 0; st >>= 1) {
    if (m < st) red[m] = fmaxf(red[m], red[m + st]);
    __syncthreads();
  }
  float mx = red[0]; __syncthreads();
  float ex = __expf(d - mx);
  red[m] = ex; __syncthreads();
  for (int st = 128; st > 0; st >>= 1) {
    if (m < st) red[m] += red[m + st];
    __syncthreads();
  }
  sup[n * 256 + m] = (half_t)(ex / red[0]);
}

// Wn (per-node weight pool) written directly in fragment-packed layout:
// [node][cheb][kt][nt][lane][16];  Wp layout [ED,2,Ci,Co].
__global__ void k_wn(const float* __restrict__ Edyn, const float* __restrict__ Wp,
                     half_t* __restrict__ Wn, int Ci, int Co) {
  int KT = Ci >> 5, NT = Co >> 4;
  long total = 256L * 2 * KT * NT * 512;
  GRID_STRIDE(i, total) {
    int e = (int)(i & 15);
    int lane = (int)((i >> 4) & 31);
    long tile = i >> 9;
    int nt = (int)(tile % NT); tile /= NT;
    int kt = (int)(tile % KT); tile /= KT;
    int cheb = (int)(tile & 1);
    int node = (int)(tile >> 1);
    int k = (kt << 5) + ((lane >> 4) << 4) + e;
    int o = (nt << 4) + (lane & 15);
    float s = 0.f;
    for (int ed = 0; ed < 10; ++ed)
      s += Edyn[node * 10 + ed] * Wp[(((long)ed * 2 + cheb) * Ci + k) * Co + o];
    Wn[i] = (half_t)s;
  }
}

__global__ void k_eb(const float* __restrict__ Edyn, const float* __restrict__ bp,
                     float* __restrict__ Eb, int Co) {
  long total = 256L * Co;
  GRID_STRIDE(i, total) {
    long n = i / Co;
    int o = (int)(i % Co);
    float s = 0.f;
    for (int e = 0; e < 10; ++e) s += Edyn[n * 10 + e] * bp[e * Co + o];
    Eb[i] = s;
  }
}

__global__ void k_slice_t(const half_t* fd, half_t* net, int t) {
  long total = 16L * 256 * 64;
  GRID_STRIDE(i, total) {
    long b = i / (256 * 64);
    long r = i % (256 * 64);
    net[i] = fd[(b * 20 + t) * (long)(256 * 64) + r];
  }
}

// packed-B index for the [B][K=256][N=128] concat activations:
// xcatB[b][kt][nt][lane][16] with m=node, c=channel
DEVINL long xcatB_idx(long b, int node, int c) {
  int kt = node >> 5;
  int rem = node & 31;
  int lane = ((rem >> 4) << 4) + (c & 15);
  int e = rem & 15;
  int nt = c >> 4;
  return (((b * 8 + kt) * 8 + nt) * 512) + (lane << 4) + e;
}

// xcat = concat(x,h): writes normal layout (gconv A side) + packed (sup-GEMM B)
__global__ void k_concat(const half_t* x, const half_t* h, half_t* xcat,
                         half_t* xcatB) {
  long total = 16L * 256 * 128;
  GRID_STRIDE(i, total) {
    int c = (int)(i % 128);
    long bn = i / 128;
    int node = (int)(bn % 256);
    long b = bn / 256;
    half_t v = (c < 64) ? x[bn * 64 + c] : h[bn * 64 + (c - 64)];
    xcat[i] = v;
    xcatB[xcatB_idx(b, node, c)] = v;
  }
}

__global__ void k_rcat(const half_t* x, const half_t* zr, const half_t* h,
                       half_t* xcat, half_t* xcatB) {
  long total = 16L * 256 * 128;
  GRID_STRIDE(i, total) {
    int c = (int)(i % 128);
    long bn = i / 128;
    int node = (int)(bn % 256);
    long b = bn / 256;
    half_t v;
    if (c < 64) v = x[bn * 64 + c];
    else {
      float rv = (float)zr[bn * 128 + 64 + (c - 64)];
      float hv = (float)h[bn * 64 + (c - 64)];
      v = (half_t)(rv * hv);
    }
    xcat[i] = v;
    xcatB[xcatB_idx(b, node, c)] = v;
  }
}

__global__ void k_gru(const half_t* zr, const half_t* cand, half_t* h) {
  long total = 16L * 256 * 64;
  GRID_STRIDE(i, total) {
    int c = (int)(i % 64);
    long bn = i / 64;
    float z = (float)zr[bn * 128 + c];
    float hv = (float)h[i], cv = (float)cand[i];
    h[i] = (half_t)(z * hv + (1.f - z) * cv);
  }
}

// bilinear 2x upsample (half-pixel centers) of z[B,16,16,64] into
// upcat[B,32,32,128] ch [0,64);  ch [64,128) <- ODE trajectory slice.
__global__ void k_upcat(const half_t* __restrict__ z, const half_t* __restrict__ trajt,
                        half_t* __restrict__ upcat) {
  long total = 16L * 32 * 32 * 64;
  GRID_STRIDE(i, total) {
    int c = (int)(i & 63);
    long r = i >> 6;
    int w = (int)(r & 31); r >>= 5;
    int h = (int)(r & 31);
    long b = r >> 5;
    float sy = h * 0.5f - 0.25f, sx = w * 0.5f - 0.25f;
    int y0 = (int)floorf(sy); float fy = sy - y0;
    int x0 = (int)floorf(sx); float fx = sx - x0;
    int ya = imin(imax(y0, 0), 15), yb = imin(imax(y0 + 1, 0), 15);
    int xa = imin(imax(x0, 0), 15), xb = imin(imax(x0 + 1, 0), 15);
    const half_t* zb = z + b * (long)(256 * 64);
    float v00 = (float)zb[(ya * 16 + xa) * 64 + c];
    float v01 = (float)zb[(ya * 16 + xb) * 64 + c];
    float v10 = (float)zb[(yb * 16 + xa) * 64 + c];
    float v11 = (float)zb[(yb * 16 + xb) * 64 + c];
    float val = (1.f - fy) * ((1.f - fx) * v00 + fx * v01) +
                fy * ((1.f - fx) * v10 + fx * v11);
    long o = ((b * 32 + h) * 32 + w) * 128;
    upcat[o + c] = (half_t)val;
    upcat[o + 64 + c] = trajt[(b * 1024 + (long)h * 32 + w) * 64 + c];
  }
}

__global__ void k_sinit(const half_t* fr, float* s, half_t* s16) {
  long total = 16L * 1024 * 64;
  GRID_STRIDE(i, total) {
    int c = (int)(i & 63);
    long r = i >> 6;
    long hw = r & 1023;
    long b = r >> 10;
    float v = (float)fr[((b * 20) * 1024 + hw) * 64 + c];
    s[i] = v;
    s16[i] = (half_t)v;
  }
}

__global__ void k_scatter(const float* x, float* out, int t) {
  long total = 16L * 1024 * 16;
  GRID_STRIDE(i, total) {
    int c = (int)(i & 15);
    long r = i >> 4;
    long hw = r & 1023;
    long b = r >> 10;
    out[((b * 19 + t) * 1024 + hw) * 16 + c] = x[i];
  }
}

__global__ void k_loss_frames(const float* __restrict__ pred,
                              const float* __restrict__ frames, float* loss) {
  const long total = 16L * 19 * 1024 * 16;
  float lc = 0.f;
  GRID_STRIDE(i, total) {
    int c = (int)(i % 16);
    long r = i / 16;
    long hw = r % 1024; r /= 1024;
    long t = r % 19;
    long b = r / 19;
    float d = pred[i] - frames[((b * 20 + t + 1) * 1024 + hw) * 16 + c];
    lc += d * d;
  }
  __shared__ float red[256];
  red[threadIdx.x] = lc; __syncthreads();
  for (int st = 128; st > 0; st >>= 1) {
    if (threadIdx.x < st) red[threadIdx.x] += red[threadIdx.x + st];
    __syncthreads();
  }
  if (threadIdx.x == 0) atomicAdd(loss, red[0] * (1.f / (float)total));
}

__global__ void k_loss_traj(const half_t* __restrict__ traj,
                            const half_t* __restrict__ fr, float* loss) {
  const long total = 19L * 16 * 1024 * 64;
  float lc = 0.f;
  GRID_STRIDE(i, total) {
    int c = (int)(i % 64);
    long r = i / 64;
    long hw = r % 1024; r /= 1024;
    long b = r % 16;
    long t = r / 16;
    float d = (float)traj[i] - (float)fr[((b * 20 + t + 1) * 1024 + hw) * 64 + c];
    lc += d * d;
  }
  __shared__ float red[256];
  red[threadIdx.x] = lc; __syncthreads();
  for (int st = 128; st > 0; st >>= 1) {
    if (threadIdx.x < st) red[threadIdx.x] += red[threadIdx.x + st];
    __syncthreads();
  }
  if (threadIdx.x == 0) atomicAdd(loss, red[0] * (1.f / (float)total));
}

// ---------------------------------------------------------------------------

static inline dim3 ewgrid(long total) {
  long b = (total + 255) / 256;
  if (b > (1L << 28)) b = 1L << 28;
  return dim3((unsigned)b);
}

extern "C" void kernel_launch(void* const* d_in, const int* in_sizes, int n_in,
                              void* d_out, int out_size, void* d_ws, size_t ws_size,
                              hipStream_t stream) {
  (void)in_sizes; (void)n_in; (void)ws_size;
  const float* frames = (const float*)d_in[0];
  const float *cE[4], *cM[4], *cWc[4], *cWg[4], *cbc[4], *cbg[4];
  for (int i = 0; i < 4; ++i) {
    int b = 2 + 6 * i;
    cE[i]  = (const float*)d_in[b + 0];
    cM[i]  = (const float*)d_in[b + 1];
    cWc[i] = (const float*)d_in[b + 2];
    cWg[i] = (const float*)d_in[b + 3];
    cbc[i] = (const float*)d_in[b + 4];
    cbg[i] = (const float*)d_in[b + 5];
  }
  const float* down_b1 = (const float*)d_in[26];
  const float* down_b2 = (const float*)d_in[27];
  const float* down_g1 = (const float*)d_in[28];
  const float* down_g2 = (const float*)d_in[29];
  const float* down_w1 = (const float*)d_in[30];
  const float* down_w2 = (const float*)d_in[31];
  const float* init_b1 = (const float*)d_in[32];
  const float* init_b2 = (const float*)d_in[33];
  const float* init_g1 = (const float*)d_in[34];
  const float* init_g2 = (const float*)d_in[35];
  const float* init_w1 = (const float*)d_in[36];
  const float* init_w2 = (const float*)d_in[37];
  const float* ode_b1  = (const float*)d_in[38];
  const float* ode_b2  = (const float*)d_in[39];
  const float* ode_w1  = (const float*)d_in[40];
  const float* ode_w2  = (const float*)d_in[41];
  const float* up_b1   = (const float*)d_in[42];
  const float* up_b2   = (const float*)d_in[43];
  const float* up_g1   = (const float*)d_in[44];
  const float* up_g2   = (const float*)d_in[45];
  const float* up_w1   = (const float*)d_in[46];
  const float* up_w2   = (const float*)d_in[47];
  const float* w_h2z   = (const float*)d_in[48];
  const float* w_last  = (const float*)d_in[49];

  // ------------------------- workspace carving ------------------------------
  char* p = (char*)d_ws;
  auto carve = [&](size_t bytes) -> char* {
    char* r = p;
    p += (bytes + 255) & ~((size_t)255);
    return r;
  };
  half_t* frames16 = (half_t*)carve(5242880UL * 2);           // [B,T,32,32,16]
  half_t* fr16     = (half_t*)carve(20971520UL * 2);          // [B*T,32,32,64]
  half_t* fd16     = (half_t*)carve(5242880UL * 2);           // [B*T,16,16,64]
  half_t* pool     = (half_t*)carve(5242880UL * 2);
  half_t* traj     = (half_t*)carve(19UL * 16 * 1024 * 64 * 2);
  float*  raw      = (float*)carve(20971520UL * 4);           // raw conv f32
  half_t* a1       = (half_t*)carve(20971520UL * 2);          // conv intermediate
  half_t* supL[4]; half_t* WnG[4]; half_t* WnC[4];
  float*  EbG[4];  float*  EbC[4]; float* Edyn[4];
  for (int i = 0; i < 4; ++i) {
    supL[i] = (half_t*)carve(65536UL * 2);
    WnG[i]  = (half_t*)carve(8388608UL * 2);   // packed [256][2][4][8][512]
    WnC[i]  = (half_t*)carve(4194304UL * 2);   // packed [256][2][4][4][512]
    EbG[i]  = (float*)carve(32768UL * 4);
    EbC[i]  = (float*)carve(16384UL * 4);
    Edyn[i] = (float*)carve(2560UL * 4);
  }
  half_t* wInit1 = (half_t*)carve(18432UL * 2);  // 9*1*4*512 (K zero-padded)
  half_t* wInit2 = (half_t*)carve(36864UL * 2);
  half_t* wDown1 = (half_t*)carve(36864UL * 2);
  half_t* wDown2 = (half_t*)carve(36864UL * 2);
  half_t* wUp1   = (half_t*)carve(73728UL * 2);
  half_t* wUp2   = (half_t*)carve(36864UL * 2);
  half_t* wH2Z   = (half_t*)carve(4096UL * 2);
  half_t* wLast  = (half_t*)carve(1024UL * 2);
  half_t* odeW1  = (half_t*)carve(8192UL * 2);
  half_t* odeW2  = (half_t*)carve(8192UL * 2);
  float*  sstate = (float*)carve(1048576UL * 4);
  half_t* s16    = (half_t*)carve(1048576UL * 2);
  half_t* hidden = (half_t*)carve(2097152UL * 2);
  half_t* hbuf   = (half_t*)carve(5UL * 262144 * 2);          // h0..h3 + zgen
  half_t* hbufs[4] = {hbuf, hbuf + 262144, hbuf + 2 * 262144, hbuf + 3 * 262144};
  half_t* zgen   = hbuf + 4 * 262144;
  half_t* netbuf = (half_t*)carve(262144UL * 2);
  half_t* xcat   = (half_t*)carve(524288UL * 2);
  half_t* xcatB  = (half_t*)carve(524288UL * 2);              // fragment-packed
  half_t* xg1    = (half_t*)carve(524288UL * 2);
  half_t* zr     = (half_t*)carve(524288UL * 2);
  half_t* cand   = (half_t*)carve(262144UL * 2);
  half_t* upcat  = (half_t*)carve(2097152UL * 2);
  half_t* u2     = (half_t*)carve(1048576UL * 2);
  float*  xtmp   = (float*)carve(262144UL * 4);
  float*  scale  = (float*)carve(64UL * 4);
  float*  shift  = (float*)carve(64UL * 4);

  const dim3 B32(32), B256(256);

  // ----------------------- weight prep (cheap, per call) --------------------
  k_f2h<<<ewgrid(5242880), B256, 0, stream>>>(frames, frames16, 5242880L);
  k_pack3x3<<<ewgrid(18432), B256, 0, stream>>>(init_w1, wInit1, 64, 16);
  k_pack3x3<<<ewgrid(36864), B256, 0, stream>>>(init_w2, wInit2, 64, 64);
  k_pack3x3<<<ewgrid(36864), B256, 0, stream>>>(down_w1, wDown1, 64, 64);
  k_pack3x3<<<ewgrid(36864), B256, 0, stream>>>(down_w2, wDown2, 64, 64);
  k_pack3x3<<<ewgrid(73728), B256, 0, stream>>>(up_w1, wUp1, 64, 128);
  k_pack3x3<<<ewgrid(36864), B256, 0, stream>>>(up_w2, wUp2, 64, 64);
  k_pack1x1<<<ewgrid(4096), B256, 0, stream>>>(w_h2z, wH2Z, 64, 64);
  k_pack1x1<<<ewgrid(1024), B256, 0, stream>>>(w_last, wLast, 16, 64);
  k_packB<<<ewgrid(8192), B256, 0, stream>>>(ode_w1, odeW1, 64, 128);
  k_packB<<<ewgrid(8192), B256, 0, stream>>>(ode_w2, odeW2, 128, 64);

  // ----- hoisted per-layer constants (data-independent in the reference) ----
  for (int l = 0; l < 4; ++l) {
    k_edyn<<<dim3(256), B32, 0, stream>>>(cE[l], cM[l], Edyn[l]);
    k_sup<<<dim3(256), B256, 0, stream>>>(Edyn[l], supL[l]);
    k_wn<<<ewgrid(8388608), B256, 0, stream>>>(Edyn[l], cWg[l], WnG[l], 128, 128);
    k_wn<<<ewgrid(4194304), B256, 0, stream>>>(Edyn[l], cWc[l], WnC[l], 128, 64);
    k_eb<<<ewgrid(32768), B256, 0, stream>>>(Edyn[l], cbg[l], EbG[l], 128);
    k_eb<<<ewgrid(16384), B256, 0, stream>>>(Edyn[l], cbc[l], EbC[l], 64);
  }

  // --------------------------- encoder convs --------------------------------
  k_conv3<<<dim3(20480, 4), B32, 0, stream>>>(frames16, wInit1, raw, 320, 32, 32, 16, 64);
  k_bnstats<<<dim3(64), B256, 0, stream>>>(raw, init_g1, init_b1, scale, shift, 327680L, 64);
  k_bnact<<<ewgrid(20971520), B256, 0, stream>>>(raw, scale, shift, a1, 20971520L, 64);
  k_conv3<<<dim3(20480, 4), B32, 0, stream>>>(a1, wInit2, raw, 320, 32, 32, 64, 64);
  k_bnstats<<<dim3(64), B256, 0, stream>>>(raw, init_g2, init_b2, scale, shift, 327680L, 64);
  k_bnact<<<ewgrid(20971520), B256, 0, stream>>>(raw, scale, shift, fr16, 20971520L, 64);
  k_maxpool<<<ewgrid(5242880), B256, 0, stream>>>(fr16, pool, 320, 32, 32, 64);
  k_conv3<<<dim3(5120, 4), B32, 0, stream>>>(pool, wDown1, raw, 320, 16, 16, 64, 64);
  k_bnstats<<<dim3(64), B256, 0, stream>>>(raw, down_g1, down_b1, scale, shift, 81920L, 64);
  k_bnact<<<ewgrid(5242880), B256, 0, stream>>>(raw, scale, shift, a1, 5242880L, 64);
  k_conv3<<<dim3(5120, 4), B32, 0, stream>>>(a1, wDown2, raw, 320, 16, 16, 64, 64);
  k_bnstats<<<dim3(64), B256, 0, stream>>>(raw, down_g2, down_b2, scale, shift, 81920L, 64);
  k_bnact<<<ewgrid(5242880), B256, 0, stream>>>(raw, scale, shift, fd16, 5242880L, 64);

  // ----------------------------- neural ODE ---------------------------------
  k_sinit<<<ewgrid(1048576), B256, 0, stream>>>(fr16, sstate, s16);
  const float dt = 1.f / 19.f;
  for (int st = 0; st < 19; ++st) {
    // hidden = tanh(s @ w1 + b1)   [16384x64]@[64x128]
    k_gemm<<<dim3(1024, 8, 1), B32, 0, stream>>>(s16, odeW1, ode_b1, hidden,
                                                 128, 64, 64, 128,
                                                 0L, 0L, 0L, /*tanh*/3, 1);
    // s += dt*(hidden @ w2 + b2), record trajectory
    k_ode2<<<dim3(1024, 4, 1), B32, 0, stream>>>(hidden, odeW2, ode_b2, sstate, s16,
                                                 traj + (long)st * 1048576, dt);
  }

  // ------------------------------ RNN scan ----------------------------------
  hipMemsetAsync(hbuf, 0, 5UL * 262144 * 2, stream);  // h0..h3, zgen = 0
  for (int t = 0; t < 19; ++t) {
    const half_t* net;
    if (t < 10) {
      k_slice_t<<<ewgrid(262144), B256, 0, stream>>>(fd16, netbuf, t);
      net = netbuf;
    } else {
      net = zgen;
    }
    for (int l = 0; l < 4; ++l) {
      half_t* h = hbufs[l];
      // gate: zr = sigmoid(gconv(concat(x,h)))
      k_concat<<<ewgrid(524288), B256, 0, stream>>>(net, h, xcat, xcatB);
      k_gemm<<<dim3(16, 8, 16), B32, 0, stream>>>(supL[l], xcatB, nullptr, xg1,
                                                  128, 256, 256, 128,
                                                  0L, 32768L, 32768L, 0, 1);
      k_gconv<<<dim3(256, 8), B32, 0, stream>>>(xcat, xg1, WnG[l], EbG[l], zr,
                                                128, 128, /*sigmoid*/2);
      // candidate: tanh(gconv(concat(x, r*h)))
      k_rcat<<<ewgrid(524288), B256, 0, stream>>>(net, zr, h, xcat, xcatB);
      k_gemm<<<dim3(16, 8, 16), B32, 0, stream>>>(supL[l], xcatB, nullptr, xg1,
                                                  128, 256, 256, 128,
                                                  0L, 32768L, 32768L, 0, 1);
      k_gconv<<<dim3(256, 4), B32, 0, stream>>>(xcat, xg1, WnC[l], EbC[l], cand,
                                                128, 64, /*tanh*/3);
      // GRU combine: h = z*h + (1-z)*cand
      k_gru<<<ewgrid(262144), B256, 0, stream>>>(zr, cand, h);
    }
    // z = 1x1 conv(h[3])   [4096x64]@[64x64]
    k_gemm<<<dim3(256, 4, 1), B32, 0, stream>>>(hbufs[3], wH2Z, nullptr, zgen,
                                                64, 64, 64, 64,
                                                0L, 0L, 0L, 0, 1);
    // bilinear upsample + concat with ODE future slice
    k_upcat<<<ewgrid(1048576), B256, 0, stream>>>(zgen, traj + (long)t * 1048576, upcat);
    // up double_conv on 16 x 32x32
    k_conv3<<<dim3(1024, 4), B32, 0, stream>>>(upcat, wUp1, raw, 16, 32, 32, 128, 64);
    k_bnstats<<<dim3(64), B256, 0, stream>>>(raw, up_g1, up_b1, scale, shift, 16384L, 64);
    k_bnact<<<ewgrid(1048576), B256, 0, stream>>>(raw, scale, shift, a1, 1048576L, 64);
    k_conv3<<<dim3(1024, 4), B32, 0, stream>>>(a1, wUp2, raw, 16, 32, 32, 64, 64);
    k_bnstats<<<dim3(64), B256, 0, stream>>>(raw, up_g2, up_b2, scale, shift, 16384L, 64);
    k_bnact<<<ewgrid(1048576), B256, 0, stream>>>(raw, scale, shift, u2, 1048576L, 64);
    // x_gen = 1x1 conv to FC=16   [16384x64]@[64x16]
    k_gemm<<<dim3(1024, 1, 1), B32, 0, stream>>>(u2, wLast, nullptr, xtmp,
                                                 16, 64, 64, 16,
                                                 0L, 0L, 0L, 0, 0);
    k_scatter<<<ewgrid(262144), B256, 0, stream>>>(xtmp, (float*)d_out, t);
  }

  // -------------------------------- loss ------------------------------------
  float* lossp = (float*)d_out + (out_size - 1);
  hipMemsetAsync(lossp, 0, sizeof(float), stream);
  k_loss_frames<<<dim3(4096), B256, 0, stream>>>((const float*)d_out, frames, lossp);
  k_loss_traj<<<dim3(4096), B256, 0, stream>>>(traj, fr16, lossp);
}
